// OptimizedAttention_20641612824806
// MI455X (gfx1250) — matrix operand order repaired and checked by reading, not compiled, not run on hardware
//
#include <hip/hip_runtime.h>
#include <hip/hip_bf16.h>

// ---------- types ----------
typedef __attribute__((ext_vector_type(16))) _Float16 v16h;
typedef __attribute__((ext_vector_type(8)))  _Float16 v8h;
typedef __attribute__((ext_vector_type(8)))  float    v8f;
typedef __attribute__((ext_vector_type(4)))  int      v4i;

#define B_  2
#define S_  2048
#define D_  2048
#define H_  16
#define DH_ 128
#define NQKV_ 6144   // 3*H*DH

// ---------- CDNA5 async global->LDS path (ASYNCcnt), with safe fallback ----------
#if defined(__gfx1250__) && __has_builtin(__builtin_amdgcn_global_load_async_to_lds_b128)
#define ASYNC_G2L 1
#else
#define ASYNC_G2L 0
#endif

#define AS_GLOBAL __attribute__((address_space(1)))
#define AS_LOCAL  __attribute__((address_space(3)))

__device__ __forceinline__ void copy16_g2l(const _Float16* g, _Float16* l) {
#if ASYNC_G2L
  // prototype: (int4 AS1*, int4 AS3*, imm offset, imm cpol)
  __builtin_amdgcn_global_load_async_to_lds_b128(
      (AS_GLOBAL v4i*)g, (AS_LOCAL v4i*)l, 0, 0);
#else
  *(v8h*)l = *(const v8h*)g;
#endif
}

__device__ __forceinline__ void wait_async() {
#if ASYNC_G2L
#if __has_builtin(__builtin_amdgcn_s_wait_asynccnt)
  __builtin_amdgcn_s_wait_asynccnt(0);
#else
  asm volatile("s_wait_asynccnt 0x0" ::: "memory");
#endif
#endif
}

// ---------- WMMA helpers (CDNA5 wave32, 16x16x32 f16 -> f32) ----------
__device__ __forceinline__ v8f wmma16(v16h a, v16h b, v8f c) {
  return __builtin_amdgcn_wmma_f32_16x16x32_f16(false, a, false, b, (short)0, c,
                                                false, false);
}

// A-fragment: elements 0..7 = K kb..kb+7 , elements 8..15 = K kb+16..kb+23
__device__ __forceinline__ v16h afrag(const _Float16* lo, const _Float16* hi) {
  v8h l = *(const v8h*)lo;
  v8h h = *(const v8h*)hi;
  v16h r;
#pragma unroll
  for (int i = 0; i < 8; ++i) { r[i] = l[i]; r[i + 8] = h[i]; }
  return r;
}

// B-fragment: column gather (lane%16 = N, elements 0..7 = K kb..kb+7, 8..15 = K kb+16..kb+23)
__device__ __forceinline__ v16h bfrag(const _Float16* p, int ld) {
  v16h r;
#pragma unroll
  for (int i = 0; i < 8; ++i) { r[i] = p[i * ld]; r[i + 8] = p[(i + 16) * ld]; }
  return r;
}

// ---------- fp32 -> f16 conversion (vectorized, HBM stream) ----------
__global__ void cvt_f32_to_f16(const float* __restrict__ in,
                               _Float16* __restrict__ out, int n8) {
  int i = blockIdx.x * blockDim.x + threadIdx.x;
  int stride = gridDim.x * blockDim.x;
  for (; i < n8; i += stride) {
    const float4* p = (const float4*)(in + (size_t)i * 8);
    float4 a = p[0], b = p[1];
    v8h o;
    o[0] = (_Float16)a.x; o[1] = (_Float16)a.y;
    o[2] = (_Float16)a.z; o[3] = (_Float16)a.w;
    o[4] = (_Float16)b.x; o[5] = (_Float16)b.y;
    o[6] = (_Float16)b.z; o[7] = (_Float16)b.w;
    *(v8h*)(out + (size_t)i * 8) = o;
  }
}

// ---------- QKV GEMM: [4096,2048] x [2048,6144], 256x128 WG tile, 8 waves ----------
// Wave w owns rows w*32..w*32+31 x all 128 cols: 2 A-frags x 8 B-frags = 16 WMMA/k-step.
// B-frags preloaded into VGPRs so the 16 WMMAs issue as a dense chain.
__global__ void __launch_bounds__(256)
qkv_gemm(const _Float16* __restrict__ A, const _Float16* __restrict__ Bm,
         _Float16* __restrict__ qh, _Float16* __restrict__ kh,
         _Float16* __restrict__ vh) {
  __shared__ _Float16 As[256][48];    // stride 96B (16B multiple)
  __shared__ _Float16 Bs[32][144];    // stride 288B
  const int t = threadIdx.x;
  const int wave = t >> 5, lane = t & 31;
  const int ls = lane & 15, hi = lane >> 4;
  const int m0 = blockIdx.y * 256, n0 = blockIdx.x * 128;
  v8f acc[2][8] = {};
  for (int kc = 0; kc < D_; kc += 32) {
#pragma unroll
    for (int i = 0; i < 4; ++i) {                   // A: 256 rows x 4 chunks
      int id = t + 256 * i;
      int ar = id >> 2, ac = id & 3;
      copy16_g2l(&A[(size_t)(m0 + ar) * D_ + kc + ac * 8], &As[ar][ac * 8]);
    }
#pragma unroll
    for (int i = 0; i < 2; ++i) {                   // B: 32 rows x 16 chunks
      int id = t + 256 * i;
      int br = id >> 4, bc = id & 15;
      copy16_g2l(&Bm[(size_t)(kc + br) * NQKV_ + n0 + bc * 8], &Bs[br][bc * 8]);
    }
    if (kc + 32 < D_) {
      __builtin_prefetch(&A[(size_t)(m0 + (t >> 2)) * D_ + kc + 32], 0, 1);
      __builtin_prefetch(&Bm[(size_t)(kc + 32 + (t >> 4)) * NQKV_ + n0], 0, 1);
    }
    wait_async();
    __syncthreads();
    const int kb = hi ? 8 : 0;
    v16h af0 = afrag(&As[wave * 32 + ls][kb], &As[wave * 32 + ls][kb + 16]);
    v16h af1 =
        afrag(&As[wave * 32 + 16 + ls][kb], &As[wave * 32 + 16 + ls][kb + 16]);
    v16h bf[8];
#pragma unroll
    for (int tt = 0; tt < 8; ++tt) bf[tt] = bfrag(&Bs[8 * hi][tt * 16 + ls], 144);
#pragma unroll
    for (int tt = 0; tt < 8; ++tt) {
      acc[0][tt] = wmma16(af0, bf[tt], acc[0][tt]);
      acc[1][tt] = wmma16(af1, bf[tt], acc[1][tt]);
    }
    __syncthreads();
  }
  // epilogue: scatter into q/k/v [ (b*16+h)*2048 + s ][128]
#pragma unroll
  for (int rb = 0; rb < 2; ++rb)
#pragma unroll
    for (int tt = 0; tt < 8; ++tt)
#pragma unroll
      for (int r = 0; r < 8; ++r) {
        int m = m0 + wave * 32 + rb * 16 + r + 8 * hi;
        int n = n0 + tt * 16 + ls;
        int b = m >> 11, s = m & 2047;
        int s3 = n >> 11, rem = n & 2047;
        int h = rem >> 7, d = rem & 127;
        _Float16* dst = (s3 == 0) ? qh : (s3 == 1) ? kh : vh;
        dst[(((size_t)(b * H_ + h)) * S_ + s) * DH_ + d] =
            (_Float16)acc[rb][tt][r];
      }
}

// ---------- RoPE (in place on f16 [B,H,S,Dh]); scale folds 1/sqrt(Dh) into Q ----------
__global__ void rope_kernel(_Float16* __restrict__ buf, float scale) {
  int idx = blockIdx.x * blockDim.x + threadIdx.x;   // B*H*S*64 threads
  int i = idx & 63;
  int s = (idx >> 6) & (S_ - 1);
  int bh = idx >> 17;
  size_t base = ((size_t)bh * S_ + s) * DH_;
  float ang = (float)s * __powf(10000.0f, -(float)i * (1.0f / 64.0f));
  float cs = __cosf(ang), sn = __sinf(ang);
  float x0 = (float)buf[base + i];
  float x1 = (float)buf[base + i + 64];
  buf[base + i]      = (_Float16)((x0 * cs - x1 * sn) * scale);
  buf[base + i + 64] = (_Float16)((x1 * cs + x0 * sn) * scale);
}

// ---------- causal flash attention: one (b,h) x 64-row block per workgroup ----------
__global__ void __launch_bounds__(128)
flash_attn(const _Float16* __restrict__ qh, const _Float16* __restrict__ kh,
           const _Float16* __restrict__ vh, _Float16* __restrict__ oh) {
  __shared__ _Float16 kT[128][72];        // K^T tile [d][key], 144B stride
  __shared__ _Float16 Vs[64 * 128];       // V tile [key][d]
  __shared__ _Float16 Ps[4][16][64];      // per-wave P staging (C->A relayout)
  const int t = threadIdx.x;
  const int wave = t >> 5, lane = t & 31;
  const int ls = lane & 15, hi = lane >> 4;
  const int rowblk = blockIdx.x, bh = blockIdx.y;
  const int b = bh >> 4, h = bh & 15;
  const int m0w = rowblk * 64 + wave * 16;
  const size_t sbase = (size_t)bh * S_ * DH_;
  const _Float16* qb = qh + sbase;
  const _Float16* kb_ = kh + sbase;
  const _Float16* vb = vh + sbase;

  // Q fragments resident in VGPRs: 16 rows x 128, 4 K-chunks of 32
  v16h qf[4];
  {
    const int row = m0w + ls;
    const int kb = hi ? 8 : 0;
#pragma unroll
    for (int c = 0; c < 4; ++c) {
      const _Float16* p = qb + (size_t)row * DH_ + c * 32 + kb;
      qf[c] = afrag(p, p + 16);
    }
  }

  float m8[8], l8[8];
  v8f acc[8] = {};
#pragma unroll
  for (int r = 0; r < 8; ++r) { m8[r] = -__builtin_inff(); l8[r] = 0.f; }

  for (int nb = 0; nb <= rowblk; ++nb) {
    const int n0 = nb * 64;
    // V tile: pure linear global->LDS copy -> async DMA path
#pragma unroll
    for (int c = 0; c < 4; ++c) {
      int id = t + 128 * c;
      copy16_g2l(vb + (size_t)n0 * DH_ + id * 8, &Vs[id * 8]);
    }
    // K block loaded transposed into LDS (manual: transpose needs VGPR bounce)
#pragma unroll
    for (int c = 0; c < 8; ++c) {
      int id = t + 128 * c;
      int key = id >> 4, d0 = (id & 15) * 8;
      v8h kv = *(const v8h*)(kb_ + (size_t)(n0 + key) * DH_ + d0);
#pragma unroll
      for (int e = 0; e < 8; ++e) kT[d0 + e][key] = kv[e];
    }
    wait_async();
    __syncthreads();

    // scores S = Q K^T  (scale already folded into Q); preload B-frags per tile
    v8f sc[4];
#pragma unroll
    for (int tt = 0; tt < 4; ++tt) {
      v16h bf[4];
#pragma unroll
      for (int c = 0; c < 4; ++c) bf[c] = bfrag(&kT[c * 32 + 8 * hi][tt * 16 + ls], 72);
      v8f s4 = {};
#pragma unroll
      for (int c = 0; c < 4; ++c) s4 = wmma16(qf[c], bf[c], s4);
      const int key = n0 + tt * 16 + ls;
#pragma unroll
      for (int r = 0; r < 8; ++r) {
        int qi = m0w + r + 8 * hi;
        if (key > qi) s4[r] = -__builtin_inff();
      }
      sc[tt] = s4;
    }

    // online softmax; row M = r + 8*hi lives in a fixed 16-lane half-group
#pragma unroll
    for (int r = 0; r < 8; ++r) {
      float v = fmaxf(fmaxf(sc[0][r], sc[1][r]), fmaxf(sc[2][r], sc[3][r]));
#pragma unroll
      for (int off = 1; off < 16; off <<= 1) v = fmaxf(v, __shfl_xor(v, off, 32));
      float mn = fmaxf(m8[r], v);
      float corr = __expf(m8[r] - mn);
      float rs = 0.f;
#pragma unroll
      for (int tt = 0; tt < 4; ++tt) {
        float p = __expf(sc[tt][r] - mn);
        sc[tt][r] = p;
        rs += p;
      }
#pragma unroll
      for (int off = 1; off < 16; off <<= 1) rs += __shfl_xor(rs, off, 32);
      l8[r] = l8[r] * corr + rs;
      m8[r] = mn;
#pragma unroll
      for (int dt = 0; dt < 8; ++dt) acc[dt][r] *= corr;
    }

    // stage P through per-wave LDS to get A-fragment layout
#pragma unroll
    for (int tt = 0; tt < 4; ++tt)
#pragma unroll
      for (int r = 0; r < 8; ++r)
        Ps[wave][r + 8 * hi][tt * 16 + ls] = (_Float16)sc[tt][r];

    // O += P V
#pragma unroll
    for (int kc = 0; kc < 2; ++kc) {
      const int kb2 = (hi ? 8 : 0) + 32 * kc;
      v16h pa = afrag(&Ps[wave][ls][kb2], &Ps[wave][ls][kb2 + 16]);
#pragma unroll
      for (int dt = 0; dt < 8; ++dt) {
        v16h bf = bfrag(&Vs[(32 * kc + 8 * hi) * 128 + dt * 16 + ls], 128);
        acc[dt] = wmma16(pa, bf, acc[dt]);
      }
    }
    __syncthreads();
  }

  // epilogue: normalize and store into attn_out [B,S,H*Dh] (f16)
#pragma unroll
  for (int dt = 0; dt < 8; ++dt)
#pragma unroll
    for (int r = 0; r < 8; ++r) {
      int srow = m0w + r + 8 * hi;
      float o = acc[dt][r] / l8[r];
      oh[((size_t)b * S_ + srow) * D_ + h * DH_ + dt * 16 + ls] = (_Float16)o;
    }
}

// ---------- output projection: attn [4096,2048] x w_out [2048,2048] -> fp32 out ----------
__global__ void __launch_bounds__(256)
out_gemm(const _Float16* __restrict__ A, const _Float16* __restrict__ Bm,
         float* __restrict__ out) {
  __shared__ _Float16 As[256][48];
  __shared__ _Float16 Bs[32][144];
  const int t = threadIdx.x;
  const int wave = t >> 5, lane = t & 31;
  const int ls = lane & 15, hi = lane >> 4;
  const int m0 = blockIdx.y * 256, n0 = blockIdx.x * 128;
  v8f acc[2][8] = {};
  for (int kc = 0; kc < D_; kc += 32) {
#pragma unroll
    for (int i = 0; i < 4; ++i) {
      int id = t + 256 * i;
      int ar = id >> 2, ac = id & 3;
      copy16_g2l(&A[(size_t)(m0 + ar) * D_ + kc + ac * 8], &As[ar][ac * 8]);
    }
#pragma unroll
    for (int i = 0; i < 2; ++i) {
      int id = t + 256 * i;
      int br = id >> 4, bc = id & 15;
      copy16_g2l(&Bm[(size_t)(kc + br) * D_ + n0 + bc * 8], &Bs[br][bc * 8]);
    }
    if (kc + 32 < D_) {
      __builtin_prefetch(&A[(size_t)(m0 + (t >> 2)) * D_ + kc + 32], 0, 1);
      __builtin_prefetch(&Bm[(size_t)(kc + 32 + (t >> 4)) * D_ + n0], 0, 1);
    }
    wait_async();
    __syncthreads();
    const int kb = hi ? 8 : 0;
    v16h af0 = afrag(&As[wave * 32 + ls][kb], &As[wave * 32 + ls][kb + 16]);
    v16h af1 =
        afrag(&As[wave * 32 + 16 + ls][kb], &As[wave * 32 + 16 + ls][kb + 16]);
    v16h bf[8];
#pragma unroll
    for (int tt = 0; tt < 8; ++tt) bf[tt] = bfrag(&Bs[8 * hi][tt * 16 + ls], 144);
#pragma unroll
    for (int tt = 0; tt < 8; ++tt) {
      acc[0][tt] = wmma16(af0, bf[tt], acc[0][tt]);
      acc[1][tt] = wmma16(af1, bf[tt], acc[1][tt]);
    }
    __syncthreads();
  }
#pragma unroll
  for (int rb = 0; rb < 2; ++rb)
#pragma unroll
    for (int tt = 0; tt < 8; ++tt)
#pragma unroll
      for (int r = 0; r < 8; ++r) {
        int m = m0 + wave * 32 + rb * 16 + r + 8 * hi;
        int n = n0 + tt * 16 + ls;
        out[(size_t)m * D_ + n] = acc[rb][tt][r];
      }
}

// ---------- host-side launch ----------
extern "C" void kernel_launch(void* const* d_in, const int* in_sizes, int n_in,
                              void* d_out, int out_size, void* d_ws,
                              size_t ws_size, hipStream_t stream) {
  (void)in_sizes; (void)n_in; (void)out_size; (void)ws_size;
  const float* x     = (const float*)d_in[0];   // [2,2048,2048]
  const float* w_qkv = (const float*)d_in[1];   // [2048,6144]
  const float* w_out = (const float*)d_in[2];   // [2048,2048]
  float* out = (float*)d_out;                   // [2,2048,2048]

  const int nx = B_ * S_ * D_;        // 8,388,608
  const int nwq = D_ * NQKV_;         // 12,582,912
  const int nwo = D_ * D_;            // 4,194,304

  char* ws = (char*)d_ws;
  _Float16* xh    = (_Float16*)(ws);
  _Float16* wqkvh = (_Float16*)(ws + 16777216);
  _Float16* wouth = (_Float16*)(ws + 41943040);
  _Float16* qh    = (_Float16*)(ws + 50331648);
  _Float16* kh    = (_Float16*)(ws + 67108864);
  _Float16* vh    = (_Float16*)(ws + 83886080);
  _Float16* atth  = (_Float16*)(ws + 100663296);

  cvt_f32_to_f16<<<2048, 256, 0, stream>>>(x, xh, nx / 8);
  cvt_f32_to_f16<<<2048, 256, 0, stream>>>(w_qkv, wqkvh, nwq / 8);
  cvt_f32_to_f16<<<2048, 256, 0, stream>>>(w_out, wouth, nwo / 8);

  qkv_gemm<<<dim3(NQKV_ / 128, (B_ * S_) / 256), 256, 0, stream>>>(xh, wqkvh,
                                                                   qh, kh, vh);

  const float qscale = 0.08838834764831845f;  // 1/sqrt(128)
  rope_kernel<<<(B_ * H_ * S_ * 64) / 256, 256, 0, stream>>>(qh, qscale);
  rope_kernel<<<(B_ * H_ * S_ * 64) / 256, 256, 0, stream>>>(kh, 1.0f);

  flash_attn<<<dim3(S_ / 64, B_ * H_), 128, 0, stream>>>(qh, kh, vh, atth);

  out_gemm<<<dim3(D_ / 128, (B_ * S_) / 256), 256, 0, stream>>>(atth, wouth,
                                                                out);
}
